// AlternateConvolution_3152505995416
// MI455X (gfx1250) — matrix operand order repaired and checked by reading, not compiled, not run on hardware
//
#include <hip/hip_runtime.h>

// MI455X / gfx1250, wave32. bf16 WMMA (16x16x32) + f32 accumulate,
// double-buffered LDS pipeline, one barrier per K-step.

typedef __attribute__((ext_vector_type(16))) __bf16 v16bf;
typedef __attribute__((ext_vector_type(8)))  __bf16 v8bf;
typedef __attribute__((ext_vector_type(8)))  float  v8f;

#define NV 4096
#define NE 16384
#define OUTV 128
#define LDS_STRIDE 40          // 32 + 8 halfword pad; rows stay 16B aligned (80B)
#define TILE_HW (128 * LDS_STRIDE)

union FragU { v16bf v; uint4 q[2]; };
union BfU   { uint4 q; v8bf v; };

// ---------------------------------------------------------------------------
// dbf[e] = bf16( dot(H_e[e,:], p) ); also pass H_e through to output tuple.
// ---------------------------------------------------------------------------
__global__ void diag_kernel(const float* __restrict__ He, const float* __restrict__ p,
                            __bf16* __restrict__ dbf, float* __restrict__ HeOut) {
    int e = blockIdx.x * blockDim.x + threadIdx.x;   // 16384 threads
    float s = 0.f;
    #pragma unroll 8
    for (int j = 0; j < 64; ++j) {
        float h = He[(size_t)e * 64 + j];
        s += h * p[j];
        HeOut[(size_t)e * 64 + j] = h;
    }
    dbf[e] = (__bf16)s;
}

// ---------------------------------------------------------------------------
// T (f32, 268 MB) -> Tbf (bf16, 134 MB; fits the 192 MB L2 for GEMM1 re-reads)
// ---------------------------------------------------------------------------
__global__ void conv_kernel(const float* __restrict__ T, __bf16* __restrict__ Tbf) {
    size_t i = ((size_t)blockIdx.x * blockDim.x + threadIdx.x) * 4;
    float4 v = *(const float4*)(T + i);
    union { uint2 u; __bf16 h[4]; } o;
    o.h[0] = (__bf16)v.x; o.h[1] = (__bf16)v.y;
    o.h[2] = (__bf16)v.z; o.h[3] = (__bf16)v.w;
    *(uint2*)(Tbf + i) = o.u;
}

// ---------------------------------------------------------------------------
// HWt[n, k] = bf16( sum_j H_v[k, j] * W[j, n] )  (transposed: contiguous K).
// ---------------------------------------------------------------------------
__global__ void hw_kernel(const float* __restrict__ Hv, const float* __restrict__ W,
                          __bf16* __restrict__ HWt) {
    __shared__ float hv[128];
    int k = blockIdx.x;           // 0..4095
    int n = threadIdx.x;          // 0..127
    hv[n] = Hv[(size_t)k * 128 + n];
    __syncthreads();
    float s = 0.f;
    #pragma unroll 8
    for (int j = 0; j < 128; ++j) s += hv[j] * W[(size_t)j * 128 + n];
    HWt[(size_t)n * NV + k] = (__bf16)s;
}

// ---------------------------------------------------------------------------
// Double-buffered WMMA GEMM core: C[128x128] += A[m0+..,k] * Bt[n0+..,k]
// Optionally scales A columns by bf16 diag (packed bf16 multiplies, no LDS
// staging, no divergence). One barrier per K-step; global prefetch of the
// next tile is issued before the WMMA block so it overlaps compute.
// ---------------------------------------------------------------------------
template <bool SCALE_A>
__device__ __forceinline__ void gemm_core(
    const __bf16* __restrict__ A, size_t lda, int m0,
    const __bf16* __restrict__ Bt, size_t ldb, int n0,
    const __bf16* __restrict__ dbf, int K,
    __bf16* As, __bf16* Bs, v8f acc[2][4])
{
    const int t    = threadIdx.x;
    const int row  = t >> 1;       // 0..127
    const int seg  = t & 1;        // halfword col group 0..15 / 16..31
    const int lane = t & 31;
    const int wave = t >> 5;
    const int wm   = wave >> 1;    // 0..3 -> row offset wm*32
    const int wn   = wave & 1;     // 0..1 -> col offset wn*64
    const int lh   = lane >> 4;    // lane half
    const int l15  = lane & 15;

    const __bf16* gA = A  + (size_t)(m0 + row) * lda + seg * 16;
    const __bf16* gB = Bt + (size_t)(n0 + row) * ldb + seg * 16;
    const __bf16* gD = dbf + seg * 16;

    // Prologue: stage first tile into registers.
    uint4 ra0 = *(const uint4*)(gA);
    uint4 ra1 = *(const uint4*)(gA + 8);
    uint4 rb0 = *(const uint4*)(gB);
    uint4 rb1 = *(const uint4*)(gB + 8);
    uint4 rd0 = {}, rd1 = {};
    if (SCALE_A) { rd0 = *(const uint4*)(gD); rd1 = *(const uint4*)(gD + 8); }

    int buf = 0;
    for (int k0 = 0; k0 < K; k0 += 32) {
        // Fold diag(d) into A with packed bf16 multiplies (v_pk_mul_bf16).
        uint4 sa0 = ra0, sa1 = ra1;
        if (SCALE_A) {
            BfU a0, a1, d0, d1;
            a0.q = ra0; d0.q = rd0; a0.v = a0.v * d0.v; sa0 = a0.q;
            a1.q = ra1; d1.q = rd1; a1.v = a1.v * d1.v; sa1 = a1.q;
        }
        __bf16* Asb = As + buf * TILE_HW;
        __bf16* Bsb = Bs + buf * TILE_HW;
        *(uint4*)&Asb[row * LDS_STRIDE + seg * 16]     = sa0;
        *(uint4*)&Asb[row * LDS_STRIDE + seg * 16 + 8] = sa1;
        *(uint4*)&Bsb[row * LDS_STRIDE + seg * 16]     = rb0;
        *(uint4*)&Bsb[row * LDS_STRIDE + seg * 16 + 8] = rb1;
        __syncthreads();   // tile `buf` ready; also fences reuse of buf two steps back

        // Prefetch next tile (overlaps the WMMA block below).
        if (k0 + 32 < K) {
            ra0 = *(const uint4*)(gA + k0 + 32);
            ra1 = *(const uint4*)(gA + k0 + 40);
            rb0 = *(const uint4*)(gB + k0 + 32);
            rb1 = *(const uint4*)(gB + k0 + 40);
            if (SCALE_A) {
                rd0 = *(const uint4*)(gD + k0 + 32);
                rd1 = *(const uint4*)(gD + k0 + 40);
            }
        }

        // Fragment gather per CDNA5 16-bit WMMA VGPR layouts.
        FragU a[2], b[4];
        #pragma unroll
        for (int tr = 0; tr < 2; ++tr) {
            int r = wm * 32 + tr * 16 + l15;     // A row = lane%16
            // elems 0..7: K = lh*8 + 0..7 ; elems 8..15: K = 16 + lh*8 + 0..7
            a[tr].q[0] = *(const uint4*)&Asb[r * LDS_STRIDE + lh * 8];
            a[tr].q[1] = *(const uint4*)&Asb[r * LDS_STRIDE + 16 + lh * 8];
        }
        #pragma unroll
        for (int tc = 0; tc < 4; ++tc) {
            int c = wn * 64 + tc * 16 + l15;     // B col = lane%16
            // elems 0..7: K = lh*8 + 0..7 ; elems 8..15: K = 16 + lh*8 + 0..7
            b[tc].q[0] = *(const uint4*)&Bsb[c * LDS_STRIDE + lh * 8];
            b[tc].q[1] = *(const uint4*)&Bsb[c * LDS_STRIDE + 16 + lh * 8];
        }
        #pragma unroll
        for (int tr = 0; tr < 2; ++tr)
            #pragma unroll
            for (int tc = 0; tc < 4; ++tc)
                acc[tr][tc] = __builtin_amdgcn_wmma_f32_16x16x32_bf16(
                    false, a[tr].v, false, b[tc].v,
                    (short)0, acc[tr][tc], false, false);

        buf ^= 1;
    }
}

// ---------------------------------------------------------------------------
// GEMM1: M1 = (T*d) @ T^T, symmetric -> only blocks with by <= bx computed.
// Epilogue: diag -> 1, elementwise * adj_v, write bf16 adjA (both triangles).
// ---------------------------------------------------------------------------
__global__ __launch_bounds__(256) void gemm1_kernel(
    const __bf16* __restrict__ Tbf, const __bf16* __restrict__ dbf,
    const float* __restrict__ adjv, __bf16* __restrict__ adjA)
{
    if (blockIdx.y > blockIdx.x) return;   // symmetric: upper-triangular blocks only
    __shared__ __align__(16) __bf16 As[2 * TILE_HW];
    __shared__ __align__(16) __bf16 Bs[2 * TILE_HW];

    const int m0 = blockIdx.y * 128;
    const int n0 = blockIdx.x * 128;
    v8f acc[2][4];
    const v8f zero = {0, 0, 0, 0, 0, 0, 0, 0};
    #pragma unroll
    for (int i = 0; i < 2; ++i)
        #pragma unroll
        for (int j = 0; j < 4; ++j) acc[i][j] = zero;

    gemm_core<true>(Tbf, NE, m0, Tbf, NE, n0, dbf, NE, As, Bs, acc);

    const int lane = threadIdx.x & 31;
    const int wave = threadIdx.x >> 5;
    const int wm = wave >> 1, wn = wave & 1;
    const int lh = lane >> 4, l15 = lane & 15;
    const bool offdiag = (blockIdx.x != blockIdx.y);

    // C/D layout: VGPR i -> M = i + lh*8, N = lane & 15
    #pragma unroll
    for (int tr = 0; tr < 2; ++tr)
        #pragma unroll
        for (int tc = 0; tc < 4; ++tc)
            #pragma unroll
            for (int i = 0; i < 8; ++i) {
                int gm = m0 + wm * 32 + tr * 16 + lh * 8 + i;
                int gn = n0 + wn * 64 + tc * 16 + l15;
                float val = acc[tr][tc][i];
                float v1  = (gm == gn) ? 1.0f : val;           // eye + (1-eye)*mult
                adjA[(size_t)gm * NV + gn] =
                    (__bf16)(v1 * adjv[(size_t)gm * NV + gn]);
                if (offdiag)                                    // mirror (gm != gn)
                    adjA[(size_t)gn * NV + gm] =
                        (__bf16)(val * adjv[(size_t)gn * NV + gm]);
            }
}

// ---------------------------------------------------------------------------
// GEMM2: ret = adjA(bf16) @ HW + bias.  M=4096, N=128, K=4096. Grid 32x1.
// ---------------------------------------------------------------------------
__global__ __launch_bounds__(256) void gemm2_kernel(
    const __bf16* __restrict__ adjA, const __bf16* __restrict__ HWt,
    const float* __restrict__ bias, float* __restrict__ out)
{
    __shared__ __align__(16) __bf16 As[2 * TILE_HW];
    __shared__ __align__(16) __bf16 Bs[2 * TILE_HW];

    const int m0 = blockIdx.x * 128;
    v8f acc[2][4];
    const v8f zero = {0, 0, 0, 0, 0, 0, 0, 0};
    #pragma unroll
    for (int i = 0; i < 2; ++i)
        #pragma unroll
        for (int j = 0; j < 4; ++j) acc[i][j] = zero;

    gemm_core<false>(adjA, NV, m0, HWt, NV, 0, nullptr, NV, As, Bs, acc);

    const int lane = threadIdx.x & 31;
    const int wave = threadIdx.x >> 5;
    const int wm = wave >> 1, wn = wave & 1;
    const int lh = lane >> 4, l15 = lane & 15;

    #pragma unroll
    for (int tr = 0; tr < 2; ++tr)
        #pragma unroll
        for (int tc = 0; tc < 4; ++tc)
            #pragma unroll
            for (int i = 0; i < 8; ++i) {
                int gm = m0 + wm * 32 + tr * 16 + lh * 8 + i;
                int gn = wn * 64 + tc * 16 + l15;              // 0..127
                out[(size_t)gm * OUTV + gn] = acc[tr][tc][i] + bias[gn];
            }
}

// ---------------------------------------------------------------------------
extern "C" void kernel_launch(void* const* d_in, const int* in_sizes, int n_in,
                              void* d_out, int out_size, void* d_ws, size_t ws_size,
                              hipStream_t stream) {
    (void)in_sizes; (void)n_in; (void)out_size; (void)ws_size;
    const float* Hv   = (const float*)d_in[0];   // [4096,128]
    const float* He   = (const float*)d_in[1];   // [16384,64]
    // d_in[2] adj_e: unused in node_layer branch
    const float* adjv = (const float*)d_in[3];   // [4096,4096]
    const float* T    = (const float*)d_in[4];   // [4096,16384]
    const float* W    = (const float*)d_in[5];   // [128,128]
    const float* p    = (const float*)d_in[6];   // [1,64]
    const float* bias = (const float*)d_in[7];   // [128]

    float* ret   = (float*)d_out;                          // [4096,128]
    float* HeOut = (float*)d_out + (size_t)NV * OUTV;      // [16384,64]

    char* ws = (char*)d_ws;
    size_t off = 0;
    __bf16* Tbf  = (__bf16*)(ws + off); off += (size_t)NV * NE * 2;   // 134.2 MB
    __bf16* adjA = (__bf16*)(ws + off); off += (size_t)NV * NV * 2;   // +33.5 MB
    __bf16* HWt  = (__bf16*)(ws + off); off += (size_t)OUTV * NV * 2; // +1 MB
    __bf16* dbf  = (__bf16*)(ws + off);                                // +32 KB

    // 1) per-edge diag values (bf16) + H_e passthrough (2nd tuple output)
    diag_kernel<<<NE / 256, 256, 0, stream>>>(He, p, dbf, HeOut);

    // 2) T -> bf16 (L2-resident operand for GEMM1)
    conv_kernel<<<(int)(((size_t)NV * NE / 4) / 256), 256, 0, stream>>>(T, Tbf);

    // 3) HWt = (H_v @ W)^T in bf16
    hw_kernel<<<NV, 128, 0, stream>>>(Hv, W, HWt);

    // 4) adjusted_A = mask(T*d @ T^T) * adj_v  (triangular, ~275 GFLOP effective)
    gemm1_kernel<<<dim3(NV / 128, NV / 128), 256, 0, stream>>>(Tbf, dbf, adjv, adjA);

    // 5) ret = adjusted_A @ HW + bias
    gemm2_kernel<<<dim3(NV / 128), 256, 0, stream>>>(adjA, HWt, bias, ret);
}
